// GATLayer_27719718928465
// MI455X (gfx1250) — matrix operand order repaired and checked by reading, not compiled, not run on hardware
//
#include <hip/hip_runtime.h>
#include <hip/hip_bf16.h>

typedef __attribute__((ext_vector_type(16))) __bf16 v16bf;
typedef __attribute__((ext_vector_type(8)))  __bf16 bf16x8;
typedef __attribute__((ext_vector_type(4)))  __bf16 bf16x4;
typedef __attribute__((ext_vector_type(8)))  float  v8f;

#define B_  4
#define N_  2048
#define TD  512
#define HID 128
#define LN_EPS    1e-5f
#define NEG_SLOPE 0.01f

// ---------------------------------------------------------------------------
// Async global->LDS copies (CDNA5, ASYNCcnt) with sync fallback.
// Builtin signature (from hipcc diagnostic): param0 = vector-of-int pointer
// in global AS, param1 = LDS-side pointer, then two constant ints.
// ---------------------------------------------------------------------------
#if defined(__HIP_DEVICE_COMPILE__) && \
    __has_builtin(__builtin_amdgcn_global_load_async_to_lds_b128) && \
    __has_builtin(__builtin_amdgcn_global_load_async_to_lds_b64)
#define ASYNC_LDS 1
#else
#define ASYNC_LDS 0
#endif

typedef int v4i_ __attribute__((vector_size(16)));
typedef int v2i_ __attribute__((vector_size(8)));

#if ASYNC_LDS
typedef __attribute__((address_space(1))) v4i_ as1_v4i;
typedef __attribute__((address_space(3))) v4i_ as3_v4i;
typedef __attribute__((address_space(1))) v2i_ as1_v2i;
typedef __attribute__((address_space(3))) v2i_ as3_v2i;
#endif

__device__ __forceinline__ void cp128(void* l, const void* g) {
#if ASYNC_LDS
    __builtin_amdgcn_global_load_async_to_lds_b128(
        (as1_v4i*)(unsigned long long)g,
        (as3_v4i*)(unsigned int)(unsigned long long)l, 0, 0);
#else
    *(int4*)l = *(const int4*)g;
#endif
}
__device__ __forceinline__ void cp64(void* l, const void* g) {
#if ASYNC_LDS
    __builtin_amdgcn_global_load_async_to_lds_b64(
        (as1_v2i*)(unsigned long long)g,
        (as3_v2i*)(unsigned int)(unsigned long long)l, 0, 0);
#else
    *(long long*)l = *(const long long*)g;
#endif
}
// keep newest stage (5 async ops/wave) in flight, older stage complete
__device__ __forceinline__ void wait_async_keep5() {
#if ASYNC_LDS
    asm volatile("s_wait_asynccnt 0x5" ::: "memory");
#endif
}
__device__ __forceinline__ void wait_async_all() {
#if ASYNC_LDS
    asm volatile("s_wait_asynccnt 0x0" ::: "memory");
#endif
}

// ---------------------------------------------------------------------------
// Kernel 0a: X (B*N,512) f32 -> bf16 row-major
// ---------------------------------------------------------------------------
__global__ __launch_bounds__(256) void cvt_x_kernel(
    const float* __restrict__ X, unsigned short* __restrict__ Xbf_us)
{
    __bf16* Xbf = reinterpret_cast<__bf16*>(Xbf_us);
    size_t i = ((size_t)blockIdx.x * 256 + threadIdx.x) * 4;
    float4 v = *(const float4*)(X + i);
    bf16x4 o;
    o[0] = (__bf16)v.x; o[1] = (__bf16)v.y; o[2] = (__bf16)v.z; o[3] = (__bf16)v.w;
    *(bf16x4*)(Xbf + i) = o;
}

// ---------------------------------------------------------------------------
// Kernel 0b: W (128,512) f32 -> Wt (512,128) bf16 (transposed)
// ---------------------------------------------------------------------------
__global__ __launch_bounds__(256) void cvt_w_kernel(
    const float* __restrict__ W, unsigned short* __restrict__ Wt_us)
{
    __bf16* Wt = reinterpret_cast<__bf16*>(Wt_us);
    int e = (blockIdx.x * 256 + threadIdx.x) * 4;   // over 128*512
    int n = e >> 9, k = e & 511;
    float4 v = *(const float4*)(W + e);
    Wt[(size_t)(k + 0) * HID + n] = (__bf16)v.x;
    Wt[(size_t)(k + 1) * HID + n] = (__bf16)v.y;
    Wt[(size_t)(k + 2) * HID + n] = (__bf16)v.z;
    Wt[(size_t)(k + 3) * HID + n] = (__bf16)v.w;
}

// ---------------------------------------------------------------------------
// Kernel 1: h = LN(X @ W^T + b); s_row = h.w_row, s_col = h.w_col.
//   Double-buffered async LDS staging, K-chunk 64 (2 WMMAs/stage).
// ---------------------------------------------------------------------------
__global__ __launch_bounds__(256) void gat_fc_ln_kernel(
    const unsigned short* __restrict__ Xbf_us,   // (B*N,512) bf16
    const unsigned short* __restrict__ Wt_us,    // (512,128) bf16
    const float* __restrict__ fcb,
    const float* __restrict__ attw,              // [0..127]=w_row [128..255]=w_col
    const float* __restrict__ lng,
    const float* __restrict__ lnb,
    unsigned short* __restrict__ h_out_us,       // (B*N,128) bf16
    float* __restrict__ s_row,
    float* __restrict__ s_col)
{
    __shared__ __bf16 lds_a[2][16 * 72];    // 16x64 tile, pad stride 72
    __shared__ __bf16 lds_b[2][64 * 136];   // 64x128 tile, pad stride 136
    __shared__ float  lds_h[16 * 128];

    const __bf16* Xbf = reinterpret_cast<const __bf16*>(Xbf_us);
    const __bf16* Wt  = reinterpret_cast<const __bf16*>(Wt_us);
    __bf16* h_out     = reinterpret_cast<__bf16*>(h_out_us);

    const int tid  = threadIdx.x;
    const int wave = tid >> 5;
    const int lane = tid & 31;
    const int hi   = (lane >> 4) & 1;
    const int r    = lane & 15;
    const int col0 = wave * 16;
    const int m0   = blockIdx.x * 16;

    // each wave issues exactly 5 async ops per stage (1x b64 + 4x b128)
    auto stage = [&](int s, int buf) {
        const int k0 = s * 64;
        {   // A tile: 16 rows x 64 bf16, one b64 per thread
            int rr = tid >> 4, cc = (tid & 15) * 4;
            cp64(&lds_a[buf][rr * 72 + cc],
                 Xbf + (size_t)(m0 + rr) * TD + k0 + cc);
        }
        {   // B tile: 64 rows x 128 bf16, four b128 per thread
            int kr = tid >> 2, c0 = (tid & 3) * 32;
            const __bf16* src = Wt + (size_t)(k0 + kr) * HID + c0;
            __bf16* dst = &lds_b[buf][kr * 136 + c0];
            #pragma unroll
            for (int q = 0; q < 4; ++q) cp128(dst + q * 8, src + q * 8);
        }
    };

    v8f acc = {};
    stage(0, 0);
    for (int s = 0; s < TD / 64; ++s) {
        const int buf = s & 1;
        if (s + 1 < TD / 64) { stage(s + 1, buf ^ 1); wait_async_keep5(); }
        else                 { wait_async_all(); }
        __syncthreads();

        #pragma unroll
        for (int kc = 0; kc < 64; kc += 32) {
            const bf16x8* ap =
                (const bf16x8*)&lds_a[buf][r * 72 + kc + hi * 8];
            bf16x8 alo = ap[0];
            bf16x8 ahi = ap[2];   // +16 elements => K+16
            v16bf a;
            #pragma unroll
            for (int i = 0; i < 8; ++i) { a[i] = alo[i]; a[i + 8] = ahi[i]; }

            v16bf b;
            #pragma unroll
            for (int i = 0; i < 8; ++i) {
                b[i]     = lds_b[buf][(kc + hi * 8 + i) * 136 + col0 + r];
                b[i + 8] = lds_b[buf][(kc + 16 + hi * 8 + i) * 136 + col0 + r];
            }

            acc = __builtin_amdgcn_wmma_f32_16x16x32_bf16(false, a, false, b,
                                                          (short)0, acc, false, false);
        }
        __syncthreads();
    }

    // add fc bias; spill C tile (C layout: row = v + 8*hi, col = lane&15)
    float bias = fcb[col0 + r];
    #pragma unroll
    for (int v = 0; v < 8; ++v)
        lds_h[(v + hi * 8) * 128 + col0 + r] = acc[v] + bias;
    __syncthreads();

    // LayerNorm + s_row/s_col: wave w owns rows 2w, 2w+1; lane owns 4 cols
    for (int rr = wave * 2; rr < wave * 2 + 2; ++rr) {
        float vals[4];
        float s = 0.f, sq = 0.f;
        #pragma unroll
        for (int q = 0; q < 4; ++q) {
            float x = lds_h[rr * 128 + lane * 4 + q];
            vals[q] = x; s += x; sq += x * x;
        }
        #pragma unroll
        for (int off = 16; off > 0; off >>= 1) {
            s  += __shfl_xor(s,  off, 32);
            sq += __shfl_xor(sq, off, 32);
        }
        float mu   = s * (1.f / 128.f);
        float var  = sq * (1.f / 128.f) - mu * mu;
        float rstd = rsqrtf(var + LN_EPS);

        float sr = 0.f, sc = 0.f;
        bf16x4 o4;
        #pragma unroll
        for (int q = 0; q < 4; ++q) {
            int   c = lane * 4 + q;
            float y = (vals[q] - mu) * rstd * lng[c] + lnb[c];
            o4[q] = (__bf16)y;
            sr += y * attw[c];
            sc += y * attw[HID + c];
        }
        *reinterpret_cast<bf16x4*>(&h_out[(size_t)(m0 + rr) * HID + lane * 4]) = o4;
        #pragma unroll
        for (int off = 16; off > 0; off >>= 1) {
            sr += __shfl_xor(sr, off, 32);
            sc += __shfl_xor(sc, off, 32);
        }
        if (lane == 0) { s_row[m0 + rr] = sr; s_col[m0 + rr] = sc; }
    }
}

// ---------------------------------------------------------------------------
// Kernel 2: fused leaky-relu + online softmax + (a @ h), flash style.
//   64 j-columns per iteration: one rescale, two WMMAs.
// ---------------------------------------------------------------------------
__global__ __launch_bounds__(256) void gat_attn_kernel(
    const unsigned short* __restrict__ h_us,  // bf16 (B*N,128)
    const float* __restrict__ s_row,
    const float* __restrict__ s_col,
    const float* __restrict__ attb,
    float* __restrict__ out)                  // (B*N,128) f32
{
    __shared__ __align__(16) float srow_sh[N_];
    __shared__ float scol_sh[16];

    const int tid  = threadIdx.x;
    const int wave = tid >> 5;
    const int lane = tid & 31;
    const int hi   = (lane >> 4) & 1;
    const int r    = lane & 15;
    const int col0 = wave * 16;

    const int    bb   = blockIdx.x >> 7;           // batch
    const int    i0   = (blockIdx.x & 127) * 16;   // query row block
    const size_t base = (size_t)bb * N_;

    {   // stage s_row[b,:] with float4 copies (2048 floats)
        const float4* src = (const float4*)(s_row + base);
        float4* dst = (float4*)srow_sh;
        dst[tid]       = src[tid];
        dst[tid + 256] = src[tid + 256];
    }
    if (tid < 16) scol_sh[tid] = s_col[base + i0 + tid];
    __syncthreads();

    const float bias = attb[0];
    const float scr  = scol_sh[r];
    const __bf16* hb = reinterpret_cast<const __bf16*>(h_us) + base * HID;

    v8f   acc   = {};
    float m_row = -3.402823466e38f;   // running max for row r
    float l_row = 0.f;                // running sum for row r

    for (int j0 = 0; j0 < N_; j0 += 64) {
        if (j0 + 64 < N_)
            __builtin_prefetch(hb + (size_t)(j0 + 64 + lane) * HID + col0, 0, 0);

        // two 16x32 e-tiles in WMMA-A layout; shared max / rescale
        float ev[32];
        float mx = -3.402823466e38f;
        #pragma unroll
        for (int jc = 0; jc < 2; ++jc) {
            #pragma unroll
            for (int i = 0; i < 16; ++i) {
                int   K = jc * 32 + i + ((i >= 8) ? 8 : 0) + hi * 8;
                float e = scr + srow_sh[j0 + K] + bias;
                e = (e >= 0.f) ? e : NEG_SLOPE * e;   // leaky relu
                ev[jc * 16 + i] = e;
                mx = fmaxf(mx, e);
            }
        }
        mx = fmaxf(mx, __shfl_xor(mx, 16, 32));   // combine K-halves of row r
        float m_new = fmaxf(m_row, mx);
        float scale = __expf(m_row - m_new);

        v16bf p0, p1;
        float psum = 0.f;
        #pragma unroll
        for (int i = 0; i < 16; ++i) {
            float pv = __expf(ev[i] - m_new);
            p0[i] = (__bf16)pv;
            psum += pv;
        }
        #pragma unroll
        for (int i = 0; i < 16; ++i) {
            float pv = __expf(ev[16 + i] - m_new);
            p1[i] = (__bf16)pv;
            psum += pv;
        }
        psum += __shfl_xor(psum, 16, 32);
        l_row = l_row * scale + psum;
        m_row = m_new;

        // rescale accumulator: factor for C row (v+8*hi) lives in lane (v+8*hi)
        #pragma unroll
        for (int v = 0; v < 8; ++v)
            acc[v] *= __shfl(scale, v + hi * 8, 32);

        v16bf bf0, bf1;
        #pragma unroll
        for (int i = 0; i < 16; ++i) {
            int K = i + ((i >= 8) ? 8 : 0) + hi * 8;
            bf0[i] = hb[(size_t)(j0 + K) * HID + col0 + r];
            bf1[i] = hb[(size_t)(j0 + 32 + K) * HID + col0 + r];
        }

        acc = __builtin_amdgcn_wmma_f32_16x16x32_bf16(false, p0, false, bf0,
                                                      (short)0, acc, false, false);
        acc = __builtin_amdgcn_wmma_f32_16x16x32_bf16(false, p1, false, bf1,
                                                      (short)0, acc, false, false);
    }

    // finalize: divide by softmax denominator of each C row
    #pragma unroll
    for (int v = 0; v < 8; ++v) {
        float lv = __shfl(l_row, v + hi * 8, 32);
        out[(base + i0 + v + hi * 8) * HID + col0 + r] = acc[v] / lv;
    }
}

// ---------------------------------------------------------------------------
extern "C" void kernel_launch(void* const* d_in, const int* in_sizes, int n_in,
                              void* d_out, int out_size, void* d_ws, size_t ws_size,
                              hipStream_t stream) {
    const float* X    = (const float*)d_in[0];  // token_embedding (B,N,512)
    const float* W    = (const float*)d_in[1];  // fc_w (128,512)
    const float* fcb  = (const float*)d_in[2];  // fc_b (128)
    const float* attw = (const float*)d_in[3];  // attn_w (1,256)
    const float* attb = (const float*)d_in[4];  // attn_b (1)
    const float* lng  = (const float*)d_in[5];  // ln_g (128)
    const float* lnb  = (const float*)d_in[6];  // ln_b (128)
    float* out = (float*)d_out;

    // workspace layout (bytes)
    char* ws = (char*)d_ws;
    unsigned short* Xbf    = (unsigned short*)(ws);                       // 8 MB
    unsigned short* Wt     = (unsigned short*)(ws + (size_t)8388608);     // 128 KB
    unsigned short* h_bf16 = (unsigned short*)(ws + (size_t)8519680);     // 2 MB
    float*          s_row  = (float*)(ws + (size_t)10616832);             // 32 KB
    float*          s_col  = (float*)(ws + (size_t)10649600);             // 32 KB

    cvt_x_kernel<<<(B_ * N_ * TD) / 1024, 256, 0, stream>>>(X, Xbf);
    cvt_w_kernel<<<(HID * TD) / 1024, 256, 0, stream>>>(W, Wt);
    gat_fc_ln_kernel<<<(B_ * N_) / 16, 256, 0, stream>>>(
        Xbf, Wt, fcb, attw, lng, lnb, h_bf16, s_row, s_col);
    gat_attn_kernel<<<(B_ * N_) / 16, 256, 0, stream>>>(
        h_bf16, s_row, s_col, attb, out);
}
